// my_LSTM_83992380441382
// MI455X (gfx1250) — matrix operand order repaired and checked by reading, not compiled, not run on hardware
//
#include <hip/hip_runtime.h>
#include <math.h>

typedef __attribute__((ext_vector_type(16))) _Float16 v16h;
typedef __attribute__((ext_vector_type(8)))  _Float16 v8h;
typedef __attribute__((ext_vector_type(8)))  float    v8f;
typedef __attribute__((ext_vector_type(4)))  unsigned u32x4;
typedef __attribute__((ext_vector_type(8)))  int      i32x8;
typedef __attribute__((ext_vector_type(4)))  int      i32x4;

#define T_STEPS 4096
#define IN_DIM  1024
#define HID     1024
#define G4      (4*HID)        // 4096 gate rows
#define NWG2    64             // persistent workgroups for the scan
#define JPW     (HID/NWG2)     // 16 h-indices per workgroup

#if __has_builtin(__builtin_amdgcn_tensor_load_to_lds) && \
    __has_builtin(__builtin_amdgcn_s_wait_tensorcnt)
#define HAS_TDM 1
#else
#define HAS_TDM 0
#endif

// ---------------------------------------------------------------------------
// Kernel 0a: f32 -> f16 conversion (x and w_ih), 8 elements per thread.
// ---------------------------------------------------------------------------
__global__ __launch_bounds__(256) void k_cvt_f16(const float* __restrict__ src,
                                                 _Float16* __restrict__ dst)
{
  const size_t i = ((size_t)blockIdx.x * blockDim.x + threadIdx.x) * 8;
  const float4 a = *reinterpret_cast<const float4*>(src + i);
  const float4 b = *reinterpret_cast<const float4*>(src + i + 4);
  v8h o;
  o[0] = (_Float16)a.x; o[1] = (_Float16)a.y;
  o[2] = (_Float16)a.z; o[3] = (_Float16)a.w;
  o[4] = (_Float16)b.x; o[5] = (_Float16)b.y;
  o[6] = (_Float16)b.z; o[7] = (_Float16)b.w;
  *reinterpret_cast<v8h*>(dst + i) = o;
}

// ---------------------------------------------------------------------------
// Kernel 1: xg[t, g] = sum_k x[t,k] * w_ih[g,k] + (b_ih[g] + b_hh[g])
// f16 WMMA, f32 accumulation. Block tile 64(M) x 128(N): 8 waves, each wave
// owns one 16-wide N column and 4 M-sub-tiles, so one B fragment feeds 4
// v_wmma per K-step. Fragments are contiguous b128 loads from the f16 copies.
// ---------------------------------------------------------------------------
__global__ __launch_bounds__(256) void k_gemm_xg(
    const _Float16* __restrict__ xh, const _Float16* __restrict__ wh,
    const float* __restrict__ b_ih, const float* __restrict__ b_hh,
    float* __restrict__ xg)
{
  const int lane = threadIdx.x & 31;
  const int wave = threadIdx.x >> 5;
  const int half = lane >> 4;          // which 16-lane half
  const int l16  = lane & 15;
  const int M0 = blockIdx.y * 64;                  // 4 sub-tiles of 16 in M
  const int N0 = blockIdx.x * 128 + wave * 16;     // 16 N-columns per wave

  v8f acc0 = {}, acc1 = {}, acc2 = {}, acc3 = {};
  for (int k0 = 0; k0 < IN_DIM; k0 += 32) {
    // B (32x16): lane holds col N=l16; half 0 -> K=0..15, half 1 -> K=16..31
    // => 16 contiguous f16 (32 B)
    const v16h b = *reinterpret_cast<const v16h*>(
        wh + (size_t)(N0 + l16) * IN_DIM + k0 + 16 * half);

    v8f* accs[4] = {&acc0, &acc1, &acc2, &acc3};
#pragma unroll
    for (int s = 0; s < 4; ++s) {
      // A (16x32): lane holds row M=l16; VGPR 0-3 -> K=16*half+0..7,
      // VGPR 4-7 -> K=16*half+8..15  => two contiguous b128 loads
      const _Float16* arow =
          xh + (size_t)(M0 + s * 16 + l16) * IN_DIM + k0 + 16 * half;
      const v8h alo = *reinterpret_cast<const v8h*>(arow);
      const v8h ahi = *reinterpret_cast<const v8h*>(arow + 8);
      const v16h a = __builtin_shufflevector(alo, ahi,
          0, 1, 2, 3, 4, 5, 6, 7, 8, 9, 10, 11, 12, 13, 14, 15);
      *accs[s] = __builtin_amdgcn_wmma_f32_16x16x32_f16(
          false, a, false, b, (short)0, *accs[s], false, false);
    }
  }

  const int col = N0 + l16;
  const float bias = b_ih[col] + b_hh[col];
  v8f accv[4] = {acc0, acc1, acc2, acc3};
#pragma unroll
  for (int s = 0; s < 4; ++s)
#pragma unroll
    for (int v = 0; v < 8; ++v) {
      const int row = M0 + s * 16 + v + 8 * half;  // C/D: M = v + 8*half
      xg[(size_t)row * G4 + col] = accv[s][v] + bias;
    }
}

// ---------------------------------------------------------------------------
// Kernel 2: persistent LSTM scan. 64 WGs, each keeps its 64 rows of w_hh
// (256 KB f32) resident in LDS, fetched via 4 Tensor-Data-Mover DMAs.
// One grid barrier per step; h ping-pongs between two global buffers.
// ---------------------------------------------------------------------------
__global__ __launch_bounds__(256, 1) void k_lstm_scan(
    const float* __restrict__ w_hh, const float* __restrict__ c0,
    const float* __restrict__ xg, float* __restrict__ h_buf,
    unsigned* __restrict__ counter)
{
  extern __shared__ float smem[];
  float* lds_w = smem;                  // 64 * 1024 f32 = 256 KB
  float* lds_h = smem + 64 * HID;       // 1024
  float* lds_g = lds_h + HID;           // 64 gate pre-activations
  float* lds_c = lds_g + 64;            // 16 cell states

  const int tid = threadIdx.x;
  const int j0  = blockIdx.x * JPW;

#if HAS_TDM
  // Tensor Data Mover preload: 4 DMAs, one per gate block of 16 contiguous
  // rows (rows q*1024+j0 .. +15), each a 2-D tile 1024 x 16 f32, row stride
  // 1024 elements. EXEC is ignored by tensor ops: issue once from wave 0.
  if (tid == 0) {
    const unsigned lds_base = (unsigned)(uintptr_t)(void*)lds_w;
#pragma unroll
    for (int q = 0; q < 4; ++q) {
      const unsigned long long gaddr =
          (unsigned long long)(uintptr_t)(w_hh + (size_t)(q * HID + j0) * HID);
      u32x4 g0;
      g0[0] = 1u;                                            // count=1 (valid)
      g0[1] = lds_base + (unsigned)(q * 16 * HID * 4);       // lds_addr
      g0[2] = (unsigned)(gaddr & 0xFFFFFFFFu);               // global_addr lo
      g0[3] = (unsigned)((gaddr >> 32) & 0x01FFFFFFu)        // global_addr hi
              | (2u << 30);                                  // type=2 (image)
      i32x8 g1;
      g1[0] = 0x20000;            // data_size=2 (4 bytes); mask/flags = 0
      g1[1] = (int)(HID << 16);   // tensor_dim0 = 1024 (bits 79:48, low half)
      g1[2] = (int)(16 << 16);    // tensor_dim1 = 16    (bits 111:80)
      g1[3] = (int)(HID << 16);   // tile_dim0 = 1024    (bits 127:112)
      g1[4] = 16;                 // tile_dim1 = 16      (bits 143:128)
      g1[5] = (int)HID;           // tensor_dim0_stride = 1024 (bits 207:160)
      g1[6] = 0;
      g1[7] = 0;                  // tensor_dim1_stride = 0 (unused, 2-D tile)
      const i32x4 z4 = {0, 0, 0, 0};
#if __clang_major__ >= 23
      const i32x8 z8 = {0, 0, 0, 0, 0, 0, 0, 0};
      __builtin_amdgcn_tensor_load_to_lds(g0, g1, z4, z4, z8, 0);
#else
      __builtin_amdgcn_tensor_load_to_lds(g0, g1, z4, z4, 0);
#endif
    }
  }
  __builtin_amdgcn_s_wait_tensorcnt(0);
#else
  for (int r = 0; r < 64; ++r) {
    const int q = r >> 4, jl = r & 15;
    const float* src = w_hh + (size_t)(q * HID + j0 + jl) * HID;
    for (int k = tid; k < HID; k += 256)
      lds_w[r * HID + k] = src[k];
  }
#endif
  if (tid < JPW) lds_c[tid] = c0[j0 + tid];
  __syncthreads();

  const int r   = tid >> 2;            // 0..63 : owned gate row (local)
  const int sub = tid & 3;             // 4 lanes cooperate per row
  const int q   = r >> 4;
  const int jl  = r & 15;
  const int grow = q * HID + j0 + jl;  // global gate row
  const unsigned nwg = gridDim.x;

  for (int t = 0; t < T_STEPS; ++t) {
    const int rp = t & 1;              // read parity
    float* h_rd = h_buf + rp * HID;
    float* h_wr = h_buf + (rp ^ 1) * HID;

    // Stage h_{t-1} into LDS (agent-scope loads: skip stale L0)
    for (int k = tid; k < HID; k += 256)
      lds_h[k] = __hip_atomic_load(&h_rd[k], __ATOMIC_RELAXED,
                                   __HIP_MEMORY_SCOPE_AGENT);
    __syncthreads();

    if (sub == 0 && t + 1 < T_STEPS)
      __builtin_prefetch(&xg[(size_t)(t + 1) * G4 + grow], 0, 1);

    // 1024-long dot product from LDS, 4 lanes x 256 elements each
    float sum = 0.f;
    const float4* wr4 =
        reinterpret_cast<const float4*>(lds_w + r * HID + sub * 256);
    const float4* hr4 = reinterpret_cast<const float4*>(lds_h + sub * 256);
#pragma unroll 8
    for (int k4 = 0; k4 < 64; ++k4) {
      const float4 wv = wr4[k4], hv = hr4[k4];
      sum = fmaf(wv.x, hv.x, sum);
      sum = fmaf(wv.y, hv.y, sum);
      sum = fmaf(wv.z, hv.z, sum);
      sum = fmaf(wv.w, hv.w, sum);
    }
    sum += __shfl_xor(sum, 1, 32);
    sum += __shfl_xor(sum, 2, 32);
    if (sub == 0)
      lds_g[r] = sum + xg[(size_t)t * G4 + grow];
    __syncthreads();

    // Elementwise gate math for this WG's 16 h-indices
    if (tid < JPW) {
      const float gi = lds_g[tid];
      const float gf = lds_g[16 + tid];
      const float gg = lds_g[32 + tid];
      const float go = lds_g[48 + tid];
      const float i = 1.f / (1.f + __expf(-gi));
      const float f = 1.f / (1.f + __expf(-gf));
      const float g = tanhf(gg);
      const float o = 1.f / (1.f + __expf(-go));
      const float c = fmaf(f, lds_c[tid], i * g);
      lds_c[tid] = c;
      const float h = o * tanhf(c);
      __hip_atomic_store(&h_wr[j0 + tid], h, __ATOMIC_RELAXED,
                         __HIP_MEMORY_SCOPE_AGENT);
    }
    __threadfence();       // make h visible device-wide before signaling
    __syncthreads();

    // Grid-wide barrier: monotonic counter, one signal per WG per step
    if (tid == 0) {
      __hip_atomic_fetch_add(counter, 1u, __ATOMIC_ACQ_REL,
                             __HIP_MEMORY_SCOPE_AGENT);
      const unsigned target = (unsigned)(t + 1) * nwg;
      while (__hip_atomic_load(counter, __ATOMIC_ACQUIRE,
                               __HIP_MEMORY_SCOPE_AGENT) < target)
        __builtin_amdgcn_s_sleep(2);
    }
    __syncthreads();
  }
}

// ---------------------------------------------------------------------------
// Kernel 3: pred = tanh(h_T . w_out + b_out)
// Final h lives in parity-0 buffer ((4095+1)&1 == 0).
// ---------------------------------------------------------------------------
__global__ __launch_bounds__(256) void k_head(
    const float* __restrict__ h, const float* __restrict__ w_out,
    const float* __restrict__ b_out, float* __restrict__ out)
{
  __shared__ float red[256];
  const int tid = threadIdx.x;
  float s = 0.f;
  for (int k = tid; k < HID; k += 256)
    s = fmaf(h[k], w_out[k], s);
  red[tid] = s;
  __syncthreads();
  for (int off = 128; off > 0; off >>= 1) {
    if (tid < off) red[tid] += red[tid + off];
    __syncthreads();
  }
  if (tid == 0) out[0] = tanhf(red[0] + b_out[0]);
}

// ---------------------------------------------------------------------------
// Kernel 0: init h ping-pong buffers and the sync counter (ws is poisoned).
// ---------------------------------------------------------------------------
__global__ void k_init(const float* __restrict__ h0,
                       float* __restrict__ h_buf, unsigned* __restrict__ counter)
{
  const int i = blockIdx.x * blockDim.x + threadIdx.x;
  if (i < HID) {
    h_buf[i]       = h0[i];   // read parity for t=0
    h_buf[HID + i] = 0.f;
  }
  if (i == 0) *counter = 0u;
}

extern "C" void kernel_launch(void* const* d_in, const int* in_sizes, int n_in,
                              void* d_out, int out_size, void* d_ws, size_t ws_size,
                              hipStream_t stream)
{
  (void)in_sizes; (void)n_in; (void)out_size; (void)ws_size;
  const float* x     = (const float*)d_in[0];
  const float* h0    = (const float*)d_in[1];
  const float* c0    = (const float*)d_in[2];
  const float* w_ih  = (const float*)d_in[3];
  const float* w_hh  = (const float*)d_in[4];
  const float* b_ih  = (const float*)d_in[5];
  const float* b_hh  = (const float*)d_in[6];
  const float* w_out = (const float*)d_in[7];
  const float* b_out = (const float*)d_in[8];
  float* out = (float*)d_out;

  char* ws = (char*)d_ws;
  const size_t xg_bytes = (size_t)T_STEPS * G4 * 4;          // 64 MB
  const size_t h16_bytes = (size_t)T_STEPS * IN_DIM * 2;     // 8 MB each
  float*    xg      = (float*)ws;
  _Float16* xh      = (_Float16*)(ws + xg_bytes);
  _Float16* wh      = (_Float16*)(ws + xg_bytes + h16_bytes);
  float*    h_buf   = (float*)(ws + xg_bytes + 2 * h16_bytes);
  unsigned* counter = (unsigned*)(ws + xg_bytes + 2 * h16_bytes + 2 * HID * 4);

  k_init<<<4, 256, 0, stream>>>(h0, h_buf, counter);

  // f32 -> f16 copies of x and w_ih (4M elements each, 8 per thread)
  const int cvt_blocks = (T_STEPS * IN_DIM / 8) / 256;       // 2048
  k_cvt_f16<<<cvt_blocks, 256, 0, stream>>>(x, xh);
  k_cvt_f16<<<cvt_blocks, 256, 0, stream>>>(w_ih, wh);

  dim3 g1(G4 / 128, T_STEPS / 64);           // 32 x 64 blocks
  k_gemm_xg<<<g1, 256, 0, stream>>>(xh, wh, b_ih, b_hh, xg);

  const size_t lds_bytes = (size_t)(64 * HID + HID + 64 + 16) * sizeof(float);
  k_lstm_scan<<<NWG2, 256, lds_bytes, stream>>>(w_hh, c0, xg, h_buf, counter);

  k_head<<<1, 256, 0, stream>>>(h_buf /* parity 0 */, w_out, b_out, out);
}
